// DimeNet_45492293599702
// MI455X (gfx1250) — matrix operand order repaired and checked by reading, not compiled, not run on hardware
//
#include <hip/hip_runtime.h>
#include <hip/hip_bf16.h>

typedef __attribute__((ext_vector_type(16))) _Float16 v16h;
typedef __attribute__((ext_vector_type(2)))  _Float16 v2h;
typedef __attribute__((ext_vector_type(8)))  float    v8f;

#define H_DIM 128
#define NR_ 6
#define NS_ 7
#define NB_ 8
#define SBF_DIM 42   // NS*NR
#define WBIG 1024    // NB*H
#define BSTR 40      // LDS B panel row stride (halves); even + mult of 8 -> aligned frags

// Zeros of spherical Bessel functions j_l, l=0..6, first 6 each.
__constant__ float c_Z[7][6] = {
  { 3.14159265358979f,  6.28318530717959f,  9.42477796076938f, 12.56637061435920f, 15.70796326794900f, 18.84955592153880f },
  { 4.49340945790906f,  7.72525183693771f, 10.90412165942890f, 14.06619391283150f, 17.22075527193080f, 20.37130295928760f },
  { 5.76345919689455f,  9.09501133047635f, 12.32294097056660f, 15.51460301088670f, 18.68903635541440f, 21.85387422270980f },
  { 6.98793200050052f, 10.41711854737990f, 13.69802315324920f, 16.92362128521380f, 20.12180617454290f, 23.30424698893970f },
  { 8.18256145257149f, 11.70490715457040f, 15.03966470761650f, 18.30125595954200f, 21.52541773335290f, 24.72756554783560f },
  { 9.35581211104275f, 12.96653017277430f, 16.35470963940830f, 19.65315210182100f, 22.90455064795860f, 26.12775013722550f },
  {10.51283540809820f, 14.20739245884250f, 17.64797487016690f, 20.98346306862600f, 24.26276802538690f, 27.50786840504910f }
};

// fast SiLU: v * rcp(1 + exp(-v))  -> v_exp_f32 + v_rcp_f32 + v_mul
__device__ __forceinline__ float siluf(float v) {
  return v * __builtin_amdgcn_rcpf(1.0f + __expf(-v));
}

// envelope with p=6: 1/x - 28 x^5 + 48 x^6 - 21 x^7, gated by x<1
__device__ __forceinline__ float env_fn(float x) {
  float x2 = x * x;
  float xp = x2 * x2 * x;                       // x^5
  float r = __builtin_amdgcn_rcpf(fmaxf(x, 1e-9f)) + (-28.0f) * xp + 48.0f * xp * x + (-21.0f) * xp * x2;
  return (x < 1.0f) ? r : 0.0f;
}

// spherical bessel j_n via upward recurrence
__device__ __forceinline__ float sph_jn(int n, float x) {
  x = fmaxf(x, 1e-9f);
  float rx = __builtin_amdgcn_rcpf(x);
  float s = __sinf(x), c = __cosf(x);
  float j0 = s * rx;
  if (n == 0) return j0;
  float j1 = s * rx * rx - c * rx;
  for (int l = 1; l < n; ++l) {
    float j2 = (float)(2 * l + 1) * rx * j1 - j0;
    j0 = j1; j1 = j2;
  }
  return j1;
}

// ---------------- LDS-staged WMMA GEMM ----------------
// C[M,N] = post( A[M,K] @ B[K,N] ),  K % 32 == 0, N % 128 == 0.
// Block: 256 threads = 8 waves -> 64M x 128N tile; wave = 16M x 64N (4 WMMA accs).
// LDS: A panel row-major [64][32] f16; B panel COLUMN-major [128][BSTR] f16 so each
// lane's 16-half fragment is contiguous (2x ds_load_b128). B staging writes k-pairs
// as packed half2 (ds_store_b32 + v_cvt_pk_f16_f32).
// post: (+bias[col]) -> (silu if ACT) -> (* mul[row,col]) -> (+ add[row,col])
template <int BIAS, int ACT, int MUL, int ADD>
__global__ void __launch_bounds__(256)
gemm_wmma_kernel(const float* __restrict__ A, const float* __restrict__ B,
                 const float* __restrict__ bias, const float* __restrict__ mul,
                 const float* __restrict__ add, float* __restrict__ C,
                 int M, int N, int K)
{
  __shared__ _Float16 lA[64 * 32];      // A panel: 64 rows x 32 k
  __shared__ _Float16 lB[128 * BSTR];   // B panel: 128 cols x 32 k (padded)

  const int lane  = threadIdx.x & 31;
  const int wave  = threadIdx.x >> 5;
  const int mgrp  = wave >> 1;        // 0..3 : which 16-row slice
  const int ngrp  = wave & 1;         // 0..1 : which 64-col half
  const int mbase = blockIdx.y * 64;
  const int nb0   = blockIdx.x * 128;

  const int kaoff = (lane >> 4) << 4; // A frag: hi lane-half holds K+16..31
  const int kboff = (lane >> 4) << 4; // B frag: hi lane-half holds K+16..31
  const int l15   = lane & 15;

  v8f acc[4];
#pragma unroll
  for (int nt = 0; nt < 4; ++nt) acc[nt] = v8f{0.f,0.f,0.f,0.f,0.f,0.f,0.f,0.f};

  for (int kk = 0; kk < K; kk += 32) {
    // ---- cooperative panel staging (f32 -> f16) ----
    {
      int r  = threadIdx.x >> 2;              // 0..63
      int c0 = (threadIdx.x & 3) * 8;         // 0,8,16,24
      int gr = mbase + r; if (gr >= M) gr = M - 1;
      const float* src = A + (size_t)gr * K + kk + c0;
#pragma unroll
      for (int q = 0; q < 8; ++q) lA[r * 32 + c0 + q] = (_Float16)src[q];
    }
    {
      int k2 = (threadIdx.x >> 4) * 2;        // k-pair: 0,2,..,30
      int c0 = (threadIdx.x & 15) * 8;        // col:    0..120
      const float* s0 = B + (size_t)(kk + k2)     * N + nb0 + c0;
      const float* s1 = B + (size_t)(kk + k2 + 1) * N + nb0 + c0;
#pragma unroll
      for (int q = 0; q < 8; ++q) {
        v2h h2; h2[0] = (_Float16)s0[q]; h2[1] = (_Float16)s1[q];
        *(v2h*)(&lB[(c0 + q) * BSTR + k2]) = h2;
      }
    }
    __syncthreads();

    // ---- A fragment (reused for 4 N-tiles): 2x contiguous 8-half runs ----
    v16h af;
    {
      int arow = (mgrp * 16 + l15) * 32;
#pragma unroll
      for (int q = 0; q < 8; ++q) {
        af[q]     = lA[arow + kaoff + q];
        af[8 + q] = lA[arow + kaoff + 8 + q];
      }
    }
#pragma unroll
    for (int nt = 0; nt < 4; ++nt) {
      v16h bf;
      int bbase = (ngrp * 64 + nt * 16 + l15) * BSTR + kboff;  // 16 contiguous halves
#pragma unroll
      for (int q = 0; q < 16; ++q) bf[q] = lB[bbase + q];
      acc[nt] = __builtin_amdgcn_wmma_f32_16x16x32_f16(false, af, false, bf, (short)0, acc[nt], false, false);
    }
    __syncthreads();
  }

  // ---- epilogue (branch-free via templates) ----
#pragma unroll
  for (int nt = 0; nt < 4; ++nt) {
    int col = nb0 + ngrp * 64 + nt * 16 + l15;
#pragma unroll
    for (int g = 0; g < 8; ++g) {
      int row = mbase + mgrp * 16 + g + ((lane >> 4) << 3);
      if (row < M) {
        float v = acc[nt][g];
        if constexpr (BIAS) v += bias[col];
        if constexpr (ACT)  v = siluf(v);
        size_t o = (size_t)row * N + col;
        if constexpr (MUL) v *= mul[o];
        if constexpr (ADD) v += add[o];
        C[o] = v;
      }
    }
  }
}

// ---------------- element-wise / scatter kernels ----------------
__global__ void zero_kernel(float* p, long long n) {
  long long i = (long long)blockIdx.x * blockDim.x + threadIdx.x;
  if (i < n) p[i] = 0.0f;
}

// per-edge: dist, rbf(6), spherical radial basis (42)
__global__ void edge_kernel(const float* __restrict__ pos, const int* __restrict__ ei,
                            const int* __restrict__ ej, const float* __restrict__ freq,
                            float* __restrict__ dist, float* __restrict__ rbf,
                            float* __restrict__ radial, int E)
{
  int e = blockIdx.x * blockDim.x + threadIdx.x;
  if (e >= E) return;
  int a = ei[e], b = ej[e];
  float dx = pos[a * 3 + 0] - pos[b * 3 + 0];
  float dy = pos[a * 3 + 1] - pos[b * 3 + 1];
  float dz = pos[a * 3 + 2] - pos[b * 3 + 2];
  float dd = sqrtf(dx * dx + dy * dy + dz * dz);
  dist[e] = dd;
  float d = dd * 0.2f;                      // /CUTOFF
  float env = env_fn(d);
#pragma unroll
  for (int m = 0; m < NR_; ++m)
    rbf[(size_t)e * NR_ + m] = env * __sinf(freq[m] * d);
  for (int l = 0; l < NS_; ++l) {
#pragma unroll
    for (int m = 0; m < NR_; ++m) {
      float z   = c_Z[l][m];
      float nrm = 1.41421356237f / fabsf(sph_jn(l + 1, z));
      radial[(size_t)e * SBF_DIM + l * NR_ + m] = env * nrm * sph_jn(l, d * z);
    }
  }
}

// per-triplet: angle -> Legendre cbf, sbf = radial[idx_kj] * cbf
__global__ void triplet_sbf_kernel(const float* __restrict__ pos, const int* __restrict__ idx_i,
                                   const int* __restrict__ idx_j, const int* __restrict__ idx_k,
                                   const int* __restrict__ idx_kj, const float* __restrict__ radial,
                                   float* __restrict__ sbf, int T)
{
  int t = blockIdx.x * blockDim.x + threadIdx.x;
  if (t >= T) return;
  int ii = idx_i[t], jj = idx_j[t], kk = idx_k[t];
  float jx = pos[jj * 3 + 0] - pos[ii * 3 + 0];
  float jy = pos[jj * 3 + 1] - pos[ii * 3 + 1];
  float jz = pos[jj * 3 + 2] - pos[ii * 3 + 2];
  float kx = pos[kk * 3 + 0] - pos[ii * 3 + 0];
  float ky = pos[kk * 3 + 1] - pos[ii * 3 + 1];
  float kz = pos[kk * 3 + 2] - pos[ii * 3 + 2];
  float a  = jx * kx + jy * ky + jz * kz;
  float cx = jy * kz - jz * ky;
  float cy = jz * kx - jx * kz;
  float cz = jx * ky - jy * kx;
  float b  = sqrtf(cx * cx + cy * cy + cz * cz);
  float r  = sqrtf(a * a + b * b);
  float c  = (r > 0.0f) ? (a * __builtin_amdgcn_rcpf(r)) : 1.0f;  // cos(atan2(b,a))

  const float INV4PI = 0.0795774715459477f;
  float cb[NS_];
  cb[0] = sqrtf(1.0f * INV4PI);
  cb[1] = sqrtf(3.0f * INV4PI) * c;
  float pm1 = 1.0f, pcur = c;
  for (int l = 1; l < NS_ - 1; ++l) {
    float pn = ((float)(2 * l + 1) * c * pcur - (float)l * pm1) * __builtin_amdgcn_rcpf((float)(l + 1));
    pm1 = pcur; pcur = pn;
    cb[l + 1] = sqrtf((float)(2 * (l + 1) + 1) * INV4PI) * pn;
  }
  size_t kjr = (size_t)idx_kj[t] * SBF_DIM;
  size_t to  = (size_t)t * SBF_DIM;
  for (int l = 0; l < NS_; ++l)
#pragma unroll
    for (int m = 0; m < NR_; ++m)
      sbf[to + l * NR_ + m] = radial[kjr + l * NR_ + m] * cb[l];
}

// sb[t,jb] = sbf[t,:] @ lin_sbf_w  (T x 8, K=42)
__global__ void sb_kernel(const float* __restrict__ sbf, const float* __restrict__ w,
                          float* __restrict__ sb, int T)
{
  int id = blockIdx.x * blockDim.x + threadIdx.x;
  if (id >= T * NB_) return;
  int t = id >> 3, jb = id & 7;
  float s = 0.0f;
  for (int q = 0; q < SBF_DIM; ++q) s += sbf[(size_t)t * SBF_DIM + q] * w[q * NB_ + jb];
  sb[id] = s;
}

// out[e,h] = post( rbf[e,:6] @ w[6,128] ) ; post = +bias -> silu? -> *mul
__global__ void rbfproj_kernel(const float* __restrict__ rbf, const float* __restrict__ w,
                               const float* __restrict__ bias, const float* __restrict__ mul,
                               float* __restrict__ out, int E, int act)
{
  int id = blockIdx.x * blockDim.x + threadIdx.x;
  if (id >= E * H_DIM) return;
  int e = id >> 7, h = id & 127;
  float s = 0.0f;
#pragma unroll
  for (int m = 0; m < NR_; ++m) s += rbf[(size_t)e * NR_ + m] * w[m * H_DIM + h];
  if (bias) s += bias[h];
  if (act)  s = siluf(s);
  if (mul)  s *= mul[id];
  out[id] = s;
}

// Aemb[e, 0:384] = concat(emb[z[i]], emb[z[j]], rb_e[e])
__global__ void emb_gather_kernel(const int* __restrict__ z, const int* __restrict__ ei,
                                  const int* __restrict__ ej, const float* __restrict__ table,
                                  const float* __restrict__ rbe, float* __restrict__ Aemb, int E)
{
  int id = blockIdx.x * blockDim.x + threadIdx.x;
  if (id >= E * 384) return;
  int e = id / 384, c = id % 384;
  float v;
  if (c < 128)      v = table[(size_t)z[ei[e]] * H_DIM + c];
  else if (c < 256) v = table[(size_t)z[ej[e]] * H_DIM + (c - 128)];
  else              v = rbe[(size_t)e * H_DIM + (c - 256)];
  Aemb[id] = v;
}

// dst[idx[e], :] += src[e, :]   (width 128)
__global__ void segsum_kernel(const float* __restrict__ src, const int* __restrict__ idx,
                              float* __restrict__ dst, int E)
{
  int id = blockIdx.x * blockDim.x + threadIdx.x;
  if (id >= E * H_DIM) return;
  int e = id >> 7, c = id & 127;
  atomicAdd(&dst[(size_t)idx[e] * H_DIM + c], src[id]);
}

// h[idx_ji[t], :] += sum_jb sb[t,jb] * y[idx_kj[t], jb*128 + :]
__global__ void trip_agg_kernel(const float* __restrict__ y, const float* __restrict__ sb,
                                const int* __restrict__ idx_kj, const int* __restrict__ idx_ji,
                                float* __restrict__ h, int T)
{
  int id = blockIdx.x * blockDim.x + threadIdx.x;
  if (id >= T * H_DIM) return;
  int t = id >> 7, c = id & 127;
  const float* yr  = y + (size_t)idx_kj[t] * WBIG + c;
  const float* sbr = sb + (size_t)t * NB_;
  float s = 0.0f;
#pragma unroll
  for (int jb = 0; jb < NB_; ++jb) s += sbr[jb] * yr[jb * H_DIM];
  atomicAdd(&h[(size_t)idx_ji[t] * H_DIM + c], s);
}

// Wt[k, jb*128+n] = W[n, jb, k]   (128 x 1024)
__global__ void wtrans_kernel(const float* __restrict__ W, float* __restrict__ Wt)
{
  int id = blockIdx.x * blockDim.x + threadIdx.x;
  if (id >= H_DIM * WBIG) return;
  int k = id >> 10, c = id & 1023;
  int jb = c >> 7, n = c & 127;
  Wt[id] = W[(size_t)n * WBIG + jb * H_DIM + k];
}

// out[batch[n], :] += src[n, :]  (width 128)
__global__ void graph_scatter_kernel(const float* __restrict__ src, const int* __restrict__ batch,
                                     float* __restrict__ out, int Natoms)
{
  int id = blockIdx.x * blockDim.x + threadIdx.x;
  if (id >= Natoms * H_DIM) return;
  int n = id >> 7, c = id & 127;
  atomicAdd(&out[(size_t)batch[n] * H_DIM + c], src[id]);
}

// ---------------- host orchestration ----------------
static inline int cdiv(long long a, long long b) { return (int)((a + b - 1) / b); }

extern "C" void kernel_launch(void* const* d_in, const int* in_sizes, int n_in,
                              void* d_out, int out_size, void* d_ws, size_t ws_size,
                              hipStream_t stream)
{
  (void)n_in; (void)ws_size;
  const int*   z      = (const int*)d_in[0];
  const float* pos    = (const float*)d_in[1];
  const int*   batch  = (const int*)d_in[2];
  const int*   ei     = (const int*)d_in[3];   // i (dst)
  const int*   ej     = (const int*)d_in[4];   // j (src)
  const int*   idx_i  = (const int*)d_in[5];
  const int*   idx_j  = (const int*)d_in[6];
  const int*   idx_k  = (const int*)d_in[7];
  const int*   idx_kj = (const int*)d_in[8];
  const int*   idx_ji = (const int*)d_in[9];
  const int Natoms = in_sizes[0];
  const int E = in_sizes[3];
  const int T = in_sizes[5];

  auto p  = [&](int k) { return (const float*)d_in[k]; };
  const float* freq  = p(11);
  const float* embT  = p(12);
  const float* elr_w = p(13);
  const float* elr_b = p(14);
  const float* el_w  = p(15);
  const float* el_b  = p(16);
  auto ob = [&](int blk, int item) { return (const float*)d_in[17 + blk * 8  + item]; };
  auto ib = [&](int blk, int item) { return (const float*)d_in[49 + blk * 21 + item]; };

  // workspace carve-out (floats)
  float* w = (float*)d_ws;
  size_t off = 0;
  auto take = [&](size_t n) { float* r = w + off; off += n; return r; };
  float* dist   = take((size_t)E);
  float* rbf    = take((size_t)E * NR_);
  float* radial = take((size_t)E * SBF_DIM);
  float* sbf    = take((size_t)T * SBF_DIM);
  float* sb     = take((size_t)T * NB_);
  float* rb     = take((size_t)E * H_DIM);
  float* rb_e   = take((size_t)E * H_DIM);
  float* Aemb   = take((size_t)E * 384);
  float* xA     = take((size_t)E * H_DIM);
  float* xB     = take((size_t)E * H_DIM);
  float* hA     = take((size_t)E * H_DIM);
  float* hB     = take((size_t)E * H_DIM);
  float* t1     = take((size_t)E * H_DIM);
  float* y      = take((size_t)E * WBIG);
  float* Wt     = take((size_t)H_DIM * WBIG);
  float* gnode  = take((size_t)Natoms * H_DIM);
  float* g1     = take((size_t)Natoms * H_DIM);
  float* g2     = take((size_t)Natoms * H_DIM);

  const int TB = 256;
  auto zero = [&](float* ptr, long long n) {
    zero_kernel<<<cdiv(n, TB), TB, 0, stream>>>(ptr, n);
  };
  // GEMM dispatch over the 4 used epilogue combos; N % 128 == 0 required.
  auto gemm = [&](const float* A, const float* B, const float* bias, const float* mul,
                  const float* add, float* C, int M, int N, int K, int act) {
    dim3 grid(N / 128, cdiv(M, 64));
    if (bias && act && mul)
      gemm_wmma_kernel<1,1,1,0><<<grid, 256, 0, stream>>>(A, B, bias, mul, add, C, M, N, K);
    else if (bias && act && add)
      gemm_wmma_kernel<1,1,0,1><<<grid, 256, 0, stream>>>(A, B, bias, mul, add, C, M, N, K);
    else if (bias && act)
      gemm_wmma_kernel<1,1,0,0><<<grid, 256, 0, stream>>>(A, B, bias, mul, add, C, M, N, K);
    else
      gemm_wmma_kernel<0,0,0,0><<<grid, 256, 0, stream>>>(A, B, bias, mul, add, C, M, N, K);
  };

  // ---- geometry / basis functions ----
  zero((float*)d_out, (long long)out_size);
  edge_kernel<<<cdiv(E, TB), TB, 0, stream>>>(pos, ei, ej, freq, dist, rbf, radial, E);
  triplet_sbf_kernel<<<cdiv(T, TB), TB, 0, stream>>>(pos, idx_i, idx_j, idx_k, idx_kj, radial, sbf, T);

  // ---- embedding ----
  rbfproj_kernel<<<cdiv((long long)E * H_DIM, TB), TB, 0, stream>>>(rbf, elr_w, elr_b, nullptr, rb_e, E, 1);
  emb_gather_kernel<<<cdiv((long long)E * 384, TB), TB, 0, stream>>>(z, ei, ej, embT, rb_e, Aemb, E);
  gemm(Aemb, el_w, el_b, nullptr, nullptr, xA, E, H_DIM, 384, 1);

  auto out_block = [&](int blk, const float* x) {
    rbfproj_kernel<<<cdiv((long long)E * H_DIM, TB), TB, 0, stream>>>(rbf, ob(blk, 0), nullptr, x, t1, E, 0);
    zero(gnode, (long long)Natoms * H_DIM);
    segsum_kernel<<<cdiv((long long)E * H_DIM, TB), TB, 0, stream>>>(t1, ei, gnode, E);
    gemm(gnode, ob(blk, 1), ob(blk, 2), nullptr, nullptr, g1, Natoms, H_DIM, H_DIM, 1);
    gemm(g1,    ob(blk, 3), ob(blk, 4), nullptr, nullptr, g2, Natoms, H_DIM, H_DIM, 1);
    gemm(g2,    ob(blk, 5), ob(blk, 6), nullptr, nullptr, g1, Natoms, H_DIM, H_DIM, 1);
    gemm(g1,    ob(blk, 7), nullptr,    nullptr, nullptr, g2, Natoms, H_DIM, H_DIM, 0);
    graph_scatter_kernel<<<cdiv((long long)Natoms * H_DIM, TB), TB, 0, stream>>>(g2, batch, (float*)d_out, Natoms);
  };

  auto inter_block = [&](int blk, const float* x, float* xout) {
    rbfproj_kernel<<<cdiv((long long)E * H_DIM, TB), TB, 0, stream>>>(rbf, ib(blk, 0), nullptr, nullptr, rb, E, 0);
    sb_kernel<<<cdiv((long long)T * NB_, TB), TB, 0, stream>>>(sbf, ib(blk, 1), sb, T);
    // x_ji -> hA ; x_kj * rb -> t1
    gemm(x, ib(blk, 4), ib(blk, 5), nullptr, nullptr, hA, E, H_DIM, H_DIM, 1);
    gemm(x, ib(blk, 2), ib(blk, 3), rb,      nullptr, t1, E, H_DIM, H_DIM, 1);
    // y = x_kj @ W^T (pre-transposed) : E x 1024
    wtrans_kernel<<<cdiv((long long)H_DIM * WBIG, TB), TB, 0, stream>>>(ib(blk, 6), Wt);
    gemm(t1, Wt, nullptr, nullptr, nullptr, y, E, WBIG, H_DIM, 0);
    // h = x_ji + segment_sum(sbf-weighted gather)
    trip_agg_kernel<<<cdiv((long long)T * H_DIM, TB), TB, 0, stream>>>(y, sb, idx_kj, idx_ji, hA, T);
    // before residual (x1): hB = hA + silu(lin2(silu(lin1(hA))))
    gemm(hA, ib(blk, 7), ib(blk, 8),  nullptr, nullptr, t1, E, H_DIM, H_DIM, 1);
    gemm(t1, ib(blk, 9), ib(blk, 10), nullptr, hA,      hB, E, H_DIM, H_DIM, 1);
    // xout = silu(lin(hB)) + x
    gemm(hB, ib(blk, 11), ib(blk, 12), nullptr, x, xout, E, H_DIM, H_DIM, 1);
    // after residuals (x2)
    gemm(xout, ib(blk, 13), ib(blk, 14), nullptr, nullptr, t1, E, H_DIM, H_DIM, 1);
    gemm(t1,   ib(blk, 15), ib(blk, 16), nullptr, xout,    hA, E, H_DIM, H_DIM, 1);
    gemm(hA,   ib(blk, 17), ib(blk, 18), nullptr, nullptr, t1, E, H_DIM, H_DIM, 1);
    gemm(t1,   ib(blk, 19), ib(blk, 20), nullptr, hA,    xout, E, H_DIM, H_DIM, 1);
  };

  out_block(0, xA);
  float* xc = xA;
  float* xn = xB;
  for (int blk = 0; blk < 3; ++blk) {
    inter_block(blk, xc, xn);
    out_block(blk + 1, xn);
    float* tmp = xc; xc = xn; xn = tmp;
  }
}
